// TopKRouter_88673894793956
// MI455X (gfx1250) — matrix operand order repaired and checked by reading, not compiled, not run on hardware
//
#include <hip/hip_runtime.h>
#include <hip/hip_bf16.h>
#include <stdint.h>

// ---------------------------------------------------------------------------
// MoE Top-K router for MI455X (gfx1250, wave32, WMMA).
//   GEMM: fp32 operands split into bf16 hi/lo, 3x v_wmma_f32_16x16x32_bf16
//   (hi*hi + hi*lo + lo*hi) -> near-fp32 logits at HBM-streaming speed.
//   A-split uses truncating hi + v_perm_b32 packing: ~6 VALU per 2 elements,
//   so conversion co-executes fully under the WMMA/XDL pipeline.
//   No LDS in GEMM: X has zero cross-wave reuse; W (1MB) stays in L2/WGP$.
// ---------------------------------------------------------------------------

typedef __attribute__((ext_vector_type(16))) __bf16         v16bf;
typedef __attribute__((ext_vector_type(8)))  float          v8f;
typedef __attribute__((ext_vector_type(4)))  float          v4f;
typedef __attribute__((ext_vector_type(8)))  unsigned       v8u;

constexpr int N_TOK = 16384;
constexpr int HID   = 4096;
constexpr int NEXP  = 64;
constexpr int TOPK  = 2;
constexpr float EPS = 1e-8f;

constexpr int STAT_BLOCKS        = 256;
constexpr int WAVES_PER_BLOCK    = 8;
constexpr int STAT_WAVES         = STAT_BLOCKS * WAVES_PER_BLOCK;   // 2048
constexpr int ROWS_PER_STAT_WAVE = N_TOK / STAT_WAVES;              // 8
constexpr int PARTIAL_STRIDE     = NEXP + 1;                        // 65

__device__ __forceinline__ unsigned short f2bf_rne(float x) {
  unsigned u = __builtin_bit_cast(unsigned, x);
  return (unsigned short)((u + 0x7FFFu + ((u >> 16) & 1u)) >> 16);
}
__device__ __forceinline__ float bfbits2f(unsigned short h) {
  unsigned u = ((unsigned)h) << 16;
  return __builtin_bit_cast(float, u);
}

// ---- Kernel 1: split router weight fp32 -> bf16 hi/lo planes in d_ws ------
// One-time cost; use RNE here for best W accuracy.
__global__ void prep_w_split(const float* __restrict__ W,
                             unsigned short* __restrict__ Whi,
                             unsigned short* __restrict__ Wlo) {
  int i = blockIdx.x * blockDim.x + threadIdx.x;
  float w = W[i];
  unsigned short hb = f2bf_rne(w);
  Whi[i] = hb;
  Wlo[i] = f2bf_rne(w - bfbits2f(hb));
}

// ---- Kernel 2: router GEMM via WMMA ---------------------------------------
// Wave = 16 rows x 64 experts (4 n-tiles). A fragment built to match B:
// lanes 0-15 hold K[0..15], lanes 16-31 K[16..31].
// B fragment: col = lane&15, lanes 0-15 hold K[0..15], lanes 16-31 K[16..31].
__global__ __launch_bounds__(256)
void router_gemm_wmma(const float* __restrict__ X,
                      const unsigned short* __restrict__ Whi,
                      const unsigned short* __restrict__ Wlo,
                      float* __restrict__ logits) {
  const int  lane   = threadIdx.x & 31;
  const int  wave   = threadIdx.x >> 5;
  const int  nlane  = lane & 15;
  const bool hiHalf = lane >= 16;
  const int  m0     = (blockIdx.x * 8 + wave) * 16;
  const int  mrow   = m0 + nlane;
  const float* __restrict__ xrow = X + (size_t)mrow * HID;
  const int  sel    = hiHalf ? 16 : 0;

  v8f acc[4] = {};

  for (int kb = 0; kb < HID; kb += 32) {
    // A fragment: 16 fp32, lane l / l+16 pairs cover full 64B lines.
    v4f a0 = *(const v4f*)(xrow + kb + sel);
    v4f a1 = *(const v4f*)(xrow + kb + sel + 4);
    v4f a2 = *(const v4f*)(xrow + kb + sel + 8);
    v4f a3 = *(const v4f*)(xrow + kb + sel + 12);
    float af[16] = {a0.x, a0.y, a0.z, a0.w, a1.x, a1.y, a1.z, a1.w,
                    a2.x, a2.y, a2.z, a2.w, a3.x, a3.y, a3.z, a3.w};

    // Truncating hi/lo split; pack two bf16 (= top16 of a float) per dword
    // with a single v_perm_b32 each.  lo = x - trunc(x) is exact in fp32.
    v8u hp, lp;
#pragma unroll
    for (int i = 0; i < 8; ++i) {
      const float x0 = af[2 * i], x1 = af[2 * i + 1];
      const unsigned u0 = __builtin_bit_cast(unsigned, x0);
      const unsigned u1 = __builtin_bit_cast(unsigned, x1);
      const float l0 = x0 - __builtin_bit_cast(float, u0 & 0xFFFF0000u);
      const float l1 = x1 - __builtin_bit_cast(float, u1 & 0xFFFF0000u);
      hp[i] = __builtin_amdgcn_perm(u1, u0, 0x07060302u);
      lp[i] = __builtin_amdgcn_perm(__builtin_bit_cast(unsigned, l1),
                                    __builtin_bit_cast(unsigned, l0),
                                    0x07060302u);
    }
    const v16bf ahi = __builtin_bit_cast(v16bf, hp);
    const v16bf alo = __builtin_bit_cast(v16bf, lp);

    const int kbb = kb + (hiHalf ? 16 : 0);
#pragma unroll
    for (int t = 0; t < 4; ++t) {
      const size_t woff = (size_t)(t * 16 + nlane) * HID + kbb;
      v16bf bhi = *(const v16bf*)(Whi + woff);   // 32B aligned
      v16bf blo = *(const v16bf*)(Wlo + woff);
      acc[t] = __builtin_amdgcn_wmma_f32_16x16x32_bf16(
          false, ahi, false, bhi, (short)0, acc[t], false, false);
      acc[t] = __builtin_amdgcn_wmma_f32_16x16x32_bf16(
          false, alo, false, bhi, (short)0, acc[t], false, false);
      acc[t] = __builtin_amdgcn_wmma_f32_16x16x32_bf16(
          false, ahi, false, blo, (short)0, acc[t], false, false);
    }
  }

  // C layout: VGPR v -> (M = v + 8*hiHalf, N = lane&15)
  const int rbase = m0 + (hiHalf ? 8 : 0);
#pragma unroll
  for (int t = 0; t < 4; ++t)
#pragma unroll
    for (int v = 0; v < 8; ++v)
      logits[(size_t)(rbase + v) * NEXP + t * 16 + nlane] = acc[t][v];
}

// ---- Kernel 3: per-row top-2 / softmax / entropy / expert-load partials ---
__global__ __launch_bounds__(256)
void row_stats(const float* __restrict__ logits,
               float* __restrict__ out_idx,
               float* __restrict__ out_w,
               float* __restrict__ partials) {
  const int lane = threadIdx.x & 31;
  const int gw   = blockIdx.x * WAVES_PER_BLOCK + (threadIdx.x >> 5);

  float pA = 0.f, pB = 0.f, plp = 0.f;
  for (int i = 0; i < ROWS_PER_STAT_WAVE; ++i) {
    const int r = gw * ROWS_PER_STAT_WAVE + i;
    const float l0 = logits[(size_t)r * NEXP + lane];
    const float l1 = logits[(size_t)r * NEXP + 32 + lane];

    // full softmax over 64 experts (wave32: 2 per lane + shuffle tree)
    float m = fmaxf(l0, l1);
#pragma unroll
    for (int off = 16; off >= 1; off >>= 1) m = fmaxf(m, __shfl_xor(m, off, 32));
    float e0 = expf(l0 - m), e1 = expf(l1 - m);
    float s = e0 + e1;
#pragma unroll
    for (int off = 16; off >= 1; off >>= 1) s += __shfl_xor(s, off, 32);
    const float p0 = e0 / s, p1 = e1 / s;
    pA += p0;
    pB += p1;
    float pl = p0 * logf(p0 + EPS) + p1 * logf(p1 + EPS);
#pragma unroll
    for (int off = 16; off >= 1; off >>= 1) pl += __shfl_xor(pl, off, 32);
    plp += pl;

    // stable top-2 (value desc, index asc — matches jax.lax.top_k)
    float v1, v2;
    int   i1, i2;
    if (l0 >= l1) { v1 = l0; i1 = lane;      v2 = l1; i2 = lane + 32; }
    else          { v1 = l1; i1 = lane + 32; v2 = l0; i2 = lane;      }
#pragma unroll
    for (int off = 16; off >= 1; off >>= 1) {
      float ov1 = __shfl_xor(v1, off, 32); int oi1 = __shfl_xor(i1, off, 32);
      float ov2 = __shfl_xor(v2, off, 32); int oi2 = __shfl_xor(i2, off, 32);
      bool b1 = (ov1 > v1) || (ov1 == v1 && oi1 < i1);
      if (b1) {
        bool b2 = (v1 > ov2) || (v1 == ov2 && i1 < oi2);
        v2 = b2 ? v1 : ov2; i2 = b2 ? i1 : oi2;
        v1 = ov1; i1 = oi1;
      } else {
        bool same = (ov1 == v1) && (oi1 == i1);
        float c2v = same ? ov2 : ov1; int c2i = same ? oi2 : oi1;
        bool b2 = (v2 > c2v) || (v2 == c2v && i2 < c2i);
        v2 = b2 ? v2 : c2v; i2 = b2 ? i2 : c2i;
      }
    }
    if (lane == 0) {
      out_idx[2 * r]     = (float)i1;
      out_idx[2 * r + 1] = (float)i2;
      const float t  = expf(v2 - v1);            // softmax over top-2
      out_w[2 * r]     = 1.f / (1.f + t);
      out_w[2 * r + 1] = t / (1.f + t);
    }
  }
  float* p = partials + (size_t)gw * PARTIAL_STRIDE;
  p[lane]      = pA;     // experts 0..31
  p[32 + lane] = pB;     // experts 32..63
  if (lane == 0) p[NEXP] = plp;
}

// ---- Kernel 4: deterministic final reduction (variance ddof=1, entropy) ---
__global__ void finalize_stats(const float* __restrict__ partials,
                               float* __restrict__ out_var,
                               float* __restrict__ out_ent) {
  __shared__ float loads[NEXP];
  const int e = threadIdx.x;  // 64 threads
  float s = 0.f;
  for (int w = 0; w < STAT_WAVES; ++w)
    s += partials[(size_t)w * PARTIAL_STRIDE + e];
  loads[e] = s / (float)N_TOK;
  __syncthreads();
  if (e == 0) {
    float mean = 0.f;
    for (int k = 0; k < NEXP; ++k) mean += loads[k];
    mean /= (float)NEXP;
    float var = 0.f;
    for (int k = 0; k < NEXP; ++k) { float d = loads[k] - mean; var += d * d; }
    var /= (float)(NEXP - 1);
    float ent = 0.f;
    for (int w = 0; w < STAT_WAVES; ++w)
      ent += partials[(size_t)w * PARTIAL_STRIDE + NEXP];
    *out_var = var;
    *out_ent = -ent / (float)N_TOK;
  }
}

extern "C" void kernel_launch(void* const* d_in, const int* in_sizes, int n_in,
                              void* d_out, int out_size, void* d_ws, size_t ws_size,
                              hipStream_t stream) {
  (void)in_sizes; (void)n_in; (void)out_size; (void)ws_size;
  const float* X = (const float*)d_in[0];   // [16384, 4096]
  const float* W = (const float*)d_in[1];   // [64, 4096]

  float* out      = (float*)d_out;
  float* logits   = out;                               // [N, E]
  float* out_idx  = out + (size_t)N_TOK * NEXP;        // [N, 2] (as float)
  float* out_w    = out_idx + (size_t)N_TOK * TOPK;    // [N, 2]
  float* out_var  = out_w + (size_t)N_TOK * TOPK;      // scalar
  float* out_ent  = out_var + 1;                       // scalar

  unsigned short* Whi = (unsigned short*)d_ws;                  // E*H bf16
  unsigned short* Wlo = Whi + (size_t)NEXP * HID;               // E*H bf16
  float* partials = (float*)(Wlo + (size_t)NEXP * HID);         // 2048*65 f32

  prep_w_split<<<(NEXP * HID) / 256, 256, 0, stream>>>(W, Whi, Wlo);
  router_gemm_wmma<<<N_TOK / 128, 256, 0, stream>>>(X, Whi, Wlo, logits);
  row_stats<<<STAT_BLOCKS, 256, 0, stream>>>(logits, out_idx, out_w, partials);
  finalize_stats<<<1, NEXP, 0, stream>>>(partials, out_var, out_ent);
}
